// VectorQuantizer_7859790152052
// MI455X (gfx1250) — matrix operand order repaired and checked by reading, not compiled, not run on hardware
//
#include <hip/hip_runtime.h>
#include <hip/hip_bf16.h>
#include <math.h>

typedef _Float16 v16h __attribute__((ext_vector_type(16)));
typedef _Float16 v8h  __attribute__((ext_vector_type(8)));
typedef float    v8f  __attribute__((ext_vector_type(8)));

#define NUM_EMB   512
#define EMB_DIM   128
#define HWSZ      1024            // 32*32
#define NPIX      65536           // 64*32*32
#define OUT_ELEMS 8388608         // 64*128*32*32
#define SCORE_BIAS 512.0f         // makes all WMMA scores strictly positive

static __device__ __forceinline__ unsigned umin2(unsigned a, unsigned b) {
    return a < b ? a : b;
}

// Min over 8 fp32 scores with the 3-bit slot index packed in the mantissa LSBs.
// Scores are strictly positive -> unsigned compare == float compare, and ties
// resolve toward the lower index (matching jnp.argmin's first-min rule).
static __device__ __forceinline__ unsigned tile_min_key(const v8f& acc) {
    unsigned k0 = (__float_as_uint(acc[0]) & 0xFFFFFE00u) | 0u;
    unsigned k1 = (__float_as_uint(acc[1]) & 0xFFFFFE00u) | 1u;
    unsigned k2 = (__float_as_uint(acc[2]) & 0xFFFFFE00u) | 2u;
    unsigned k3 = (__float_as_uint(acc[3]) & 0xFFFFFE00u) | 3u;
    unsigned k4 = (__float_as_uint(acc[4]) & 0xFFFFFE00u) | 4u;
    unsigned k5 = (__float_as_uint(acc[5]) & 0xFFFFFE00u) | 5u;
    unsigned k6 = (__float_as_uint(acc[6]) & 0xFFFFFE00u) | 6u;
    unsigned k7 = (__float_as_uint(acc[7]) & 0xFFFFFE00u) | 7u;
    return umin2(umin2(umin2(k0, k1), umin2(k2, k3)),
                 umin2(umin2(k4, k5), umin2(k6, k7)));
}

// ---------------------------------------------------------------------------
// Prep: wh[k][c] = f16(-2*w[k][c]) (WMMA then emits -2*x.w directly),
//       w2[k]    = ||w_k||^2 + SCORE_BIAS (fp32 seed for the WMMA accumulator;
//                  the bias keeps every score positive for the key-pack argmin)
// ---------------------------------------------------------------------------
__global__ __launch_bounds__(128) void vq_prep(const float* __restrict__ w,
                                               _Float16* __restrict__ wh,
                                               float* __restrict__ w2) {
    const int k = blockIdx.x;
    const int c = threadIdx.x;
    float v = w[k * EMB_DIM + c];
    wh[k * EMB_DIM + c] = (_Float16)(-2.0f * v);
    float s = v * v;
    #pragma unroll
    for (int off = 16; off; off >>= 1) s += __shfl_xor(s, off, 32);
    __shared__ float red[4];
    if ((threadIdx.x & 31) == 0) red[threadIdx.x >> 5] = s;
    __syncthreads();
    if (threadIdx.x == 0) w2[k] = red[0] + red[1] + red[2] + red[3] + SCORE_BIAS;
}

// Zero the histogram + loss accumulator every launch (graph-replay safe).
__global__ __launch_bounds__(256) void vq_zero(int* __restrict__ counts,
                                               float* __restrict__ lossAcc) {
    const int t = threadIdx.x;
    counts[t]       = 0;
    counts[t + 256] = 0;
    if (t == 0) *lossAcc = 0.0f;
}

// ---------------------------------------------------------------------------
// Main: each wave owns 32 pixels (two 16-pixel B groups sharing every A
// fragment -> halves L2 codebook traffic, doubles WMMA per A-load).
// WMMA: A = 16 codes x 32 ch (f16 of -2w), B = 32 ch x 16 pixels (f16 of x),
//       C seeded with ||w||^2 + bias -> D = bias + ||w||^2 - 2 x.w  (fp32, >0)
// Lane (n = lane&15, hi = lane>>4) holds 8 codes for pixel n; per-lane packed
// u32 key-min + shfl_xor(16) merge gives the argmin per pixel.
// Epilogue: gather fp32 w[idx] rows -> NCHW output, fused loss + histogram.
// ---------------------------------------------------------------------------
__global__ __launch_bounds__(256) void vq_main(const float*    __restrict__ x,
                                               const float*    __restrict__ w,
                                               const _Float16* __restrict__ wh,
                                               const float*    __restrict__ w2,
                                               int*            __restrict__ counts,
                                               float*          __restrict__ lossAcc,
                                               float*          __restrict__ out) {
    __shared__ int s_idx[256];

    const int lane = threadIdx.x & 31;
    const int wv   = threadIdx.x >> 5;
    const int hi   = lane >> 4;     // half-wave
    const int nn   = lane & 15;     // pixel column / A-matrix row within tile

    const int  blockPix0 = blockIdx.x * 256;          // 256 pixels per block
    const int  n_img     = blockPix0 >> 10;           // image index
    const int  q0        = blockPix0 & 1023;          // offset within 32x32 plane
    const long nbase     = (long)n_img * (EMB_DIM * HWSZ);

    const int qA = q0 + wv * 32 + nn;                 // pixel group 0
    const int qB = qA + 16;                           // pixel group 1

    // ---- B fragments (pixels), fp32 -> f16, resident all sweep ----
    union U16 { v16h v; _Float16 e[16]; };
    U16 bfA[4], bfB[4];
    #pragma unroll
    for (int cc = 0; cc < 4; ++cc) {
        const int cb = cc * 32 + hi * 16;             // element e -> channel cb + e
        #pragma unroll
        for (int e = 0; e < 16; ++e) {
            bfA[cc].e[e] = (_Float16)x[nbase + (long)(cb + e) * HWSZ + qA];
            bfB[cc].e[e] = (_Float16)x[nbase + (long)(cb + e) * HWSZ + qB];
        }
    }

    unsigned best0 = 0xFFFFFFFFu;
    unsigned best1 = 0xFFFFFFFFu;
    const unsigned vkb = (unsigned)(8 * hi);          // per-half code offset

    for (int k0 = 0; k0 < NUM_EMB; k0 += 16) {
        // C init: rows m = k0 + 8*hi + r carry ||w_m||^2 + bias
        v8f acc0 = *(const v8f*)(w2 + k0 + 8 * hi);
        v8f acc1 = acc0;

        const _Float16* arow = wh + (long)(k0 + nn) * EMB_DIM;  // code row m = nn
        #pragma unroll
        for (int cc = 0; cc < 4; ++cc) {
            // A-layout (matches B): elems 0..7  = channels cc*32 + 16*hi + [0,8)
            //                       elems 8..15 = channels cc*32 + 16*hi + 8 + [0,8)
            union { v16h v; v8h h[2]; } a;
            a.h[0] = *(const v8h*)(arow + cc * 32 + 16 * hi);
            a.h[1] = *(const v8h*)(arow + cc * 32 + 16 * hi + 8);
            acc0 = __builtin_amdgcn_wmma_f32_16x16x32_f16(
                false, a.v, false, bfA[cc].v, (short)0, acc0, false, false);
            acc1 = __builtin_amdgcn_wmma_f32_16x16x32_f16(
                false, a.v, false, bfB[cc].v, (short)0, acc1, false, false);
        }
        const unsigned kb = (unsigned)k0 | vkb;       // tile's 9-bit code base
        best0 = umin2(best0, tile_min_key(acc0) | kb);
        best1 = umin2(best1, tile_min_key(acc1) | kb);
    }
    // merge the two half-wave code groups (key ties -> lower index, built-in)
    best0 = umin2(best0, (unsigned)__shfl_xor((int)best0, 16, 32));
    best1 = umin2(best1, (unsigned)__shfl_xor((int)best1, 16, 32));
    const int i0 = (int)(best0 & 0x1FFu);
    const int i1 = (int)(best1 & 0x1FFu);

    if (hi == 0) {
        s_idx[wv * 32 + nn] = i0;
        atomicAdd(counts + i0, 1);                    // 512-bin histogram
    } else {
        s_idx[wv * 32 + 16 + nn] = i1;
        atomicAdd(counts + i1, 1);
    }
    __syncthreads();

    // ---- epilogue: quantized output (fp32 codebook gather) + fused loss ----
    const int p    = threadIdx.x;                     // pixel-in-block
    const int kidx = s_idx[p];
    const float* wrow = w + (long)kidx * EMB_DIM;
    float lsum = 0.0f;
    #pragma unroll 4
    for (int c = 0; c < EMB_DIM; ++c) {
        const long off = nbase + (long)c * HWSZ + q0 + p;  // contiguous across lanes
        float qv = wrow[c];
        float xv = x[off];
        out[off] = qv;
        float d = qv - xv;
        lsum += d * d;
    }
    #pragma unroll
    for (int off = 16; off; off >>= 1) lsum += __shfl_xor(lsum, off, 32);
    if (lane == 0) atomicAdd(lossAcc, lsum);
}

// ---------------------------------------------------------------------------
// Finalize: loss = 1.25 * sum_sq / (N*C*H*W);  perplexity = exp(-sum p log(p+eps))
// ---------------------------------------------------------------------------
__global__ __launch_bounds__(512) void vq_final(const int* __restrict__ counts,
                                                const float* __restrict__ lossAcc,
                                                float* __restrict__ out) {
    __shared__ float red[16];
    const int t = threadIdx.x;
    float pb   = (float)counts[t] * (1.0f / 65536.0f);
    float term = pb * logf(pb + 1e-10f);
    #pragma unroll
    for (int off = 16; off; off >>= 1) term += __shfl_xor(term, off, 32);
    if ((t & 31) == 0) red[t >> 5] = term;
    __syncthreads();
    if (t < 32) {
        float v = (t < 16) ? red[t] : 0.0f;
        #pragma unroll
        for (int off = 8; off; off >>= 1) v += __shfl_xor(v, off, 32);
        if (t == 0) {
            out[OUT_ELEMS]     = 1.25f * (*lossAcc) * (1.0f / 8388608.0f);
            out[OUT_ELEMS + 1] = expf(-v);
        }
    }
}

extern "C" void kernel_launch(void* const* d_in, const int* in_sizes, int n_in,
                              void* d_out, int out_size, void* d_ws, size_t ws_size,
                              hipStream_t stream) {
    const float* x = (const float*)d_in[0];   // (64,128,32,32) fp32
    const float* w = (const float*)d_in[1];   // (512,128) fp32
    float* out = (float*)d_out;               // 8388608 quantized + loss + perplexity

    // workspace layout
    char*     ws      = (char*)d_ws;
    _Float16* wh      = (_Float16*)ws;                    // 131072 B: f16(-2*w)
    float*    w2      = (float*)(ws + 131072);            //   2048 B: ||w_k||^2 + bias
    int*      counts  = (int*)(ws + 131072 + 2048);       //   2048 B: histogram
    float*    lossAcc = (float*)(ws + 131072 + 4096);     //      4 B

    vq_prep <<<NUM_EMB, 128, 0, stream>>>(w, wh, w2);
    vq_zero <<<1, 256, 0, stream>>>(counts, lossAcc);
    vq_main <<<NPIX / 256, 256, 0, stream>>>(x, w, wh, w2, counts, lossAcc, out);
    vq_final<<<1, 512, 0, stream>>>(counts, lossAcc, out);
}
